// SelfAttention_56461640073376
// MI455X (gfx1250) — matrix operand order repaired and checked by reading, not compiled, not run on hardware
//
#include <hip/hip_runtime.h>
#include <hip/hip_bf16.h>
#include <math.h>

// ---------------- problem constants ----------------
#define SEQ   2048
#define HID   4096
#define NHEAD 32
#define HDIM  128          // head dim
#define QKV_N (3*HID)      // 12288

// ---------------- WMMA types ----------------
typedef __attribute__((ext_vector_type(16))) __bf16 v16bf;
typedef __attribute__((ext_vector_type(8)))  float  v8f;

union FragU { unsigned u[8]; v16bf v; };

static __device__ inline v8f wmma_bf16(v16bf a, v16bf b, v8f c) {
  // D = A(16x32 bf16) x B(32x16 bf16) + C(16x16 f32)
  return __builtin_amdgcn_wmma_f32_16x16x32_bf16(
      /*neg_a=*/false, a, /*neg_b=*/false, b,
      /*c_mod=*/(short)0, c, /*reuse_a=*/false, /*reuse_b=*/false);
}

// bf16 helpers — native hardware converts (RNE on gfx1250)
static __device__ inline unsigned short f2bf(float f) {
  union { __bf16 h; unsigned short u; } x; x.h = (__bf16)f; return x.u;
}
static __device__ inline float bf2f(unsigned short h) {
  union { unsigned u; float f; } x; x.u = ((unsigned)h) << 16;
  return x.f;
}
static __device__ inline unsigned pack2(float lo, float hi) {
  union { __bf16 h[2]; unsigned u; } x;
  x.h[0] = (__bf16)lo; x.h[1] = (__bf16)hi;
  return x.u;
}

// A-fragment K index for VGPR v, lane-half h (16-bit A 16x32 layout)
static __device__ inline int a_kidx(int v, int h) {
  return ((v < 4) ? 0 : 16) + h * 8 + (v & 3) * 2;
}
// B-fragment K index for VGPR v, lane-half h (16-bit B 32x16 layout)
static __device__ inline int b_kidx(int v, int h) {
  return ((v < 4) ? 0 : 16) + h * 8 + (v & 3) * 2;
}

// =====================================================================
// GEMM: C[M,N] = A[M,K] (f32, row-major) x W[K,N] (f32, row-major) + bias
// Block tile 128x64, BK=32 (one WMMA K-step). 8 waves in a 4x2 grid,
// each wave owns a 32x32 patch => 4 WMMA accumulators.
// A-slab is DMA'd raw f32 into LDS with GLOBAL_LOAD_ASYNC_TO_LDS_B128
// (ASYNCcnt path, no VGPR staging); B-slab is converted to bf16 and
// stored transposed so B-fragments are contiguous K-pairs.
// =====================================================================
#define BM 128
#define BN 64
#define BK 32
#define LDA_SF 36   // float stride for raw-f32 A tile (mult of 4 -> 16B rows)
#define LDB_S  40   // ushort stride for transposed bf16 B tile

template <bool OUT_BF16>
__global__ __launch_bounds__(256)
void gemm_bf16_kernel(const float* __restrict__ A, const float* __restrict__ W,
                      const float* __restrict__ bias, void* __restrict__ Cout,
                      int M, int N, int K) {
  __shared__ float          Asf[BM * LDA_SF];   // raw f32 [m][k]
  __shared__ unsigned short Bs[BN * LDB_S];     // bf16 transposed [n][k]

  const int tid  = threadIdx.x;
  const int lane = tid & 31;
  const int wave = __builtin_amdgcn_readfirstlane(tid >> 5);
  const int wm   = wave >> 1;        // 0..3
  const int wn   = wave & 1;         // 0..1
  const int m0   = blockIdx.y * BM;
  const int n0   = blockIdx.x * BN;
  const int mlow = lane & 15;
  const int half = lane >> 4;

  const v8f vzero = {0.f,0.f,0.f,0.f,0.f,0.f,0.f,0.f};
  v8f acc[2][2];
  acc[0][0] = vzero; acc[0][1] = vzero; acc[1][0] = vzero; acc[1][1] = vzero;

  const int ac = (tid & 7) * 4;      // A tile: col of float4
  const int ar = tid >> 3;           // A tile: base row (0..31)
  const int bc = (tid & 15) * 4;     // B tile: col of float4
  const int br = tid >> 4;           // B tile: base row (0..15)

  for (int k0 = 0; k0 < K; k0 += BK) {
    // ---- A slab: async DMA global -> LDS (raw f32, 16B per lane) ----
#pragma unroll
    for (int it = 0; it < 4; ++it) {
      const int r = ar + it * 32;
      const unsigned long long gp =
          (unsigned long long)(A + (size_t)(m0 + r) * K + k0 + ac);
      const unsigned lds = (unsigned)(size_t)&Asf[r * LDA_SF + ac];
      __asm__ __volatile__("global_load_async_to_lds_b128 %0, %1, off"
                           :: "v"(lds), "v"(gp) : "memory");
    }
    // ---- B slab (32x64): load, convert bf16, store transposed ----
#pragma unroll
    for (int it = 0; it < 2; ++it) {
      int r = br + it * 16;
      const float4 f = *(const float4*)(W + (size_t)(k0 + r) * N + n0 + bc);
      Bs[(bc + 0) * LDB_S + r] = f2bf(f.x);
      Bs[(bc + 1) * LDB_S + r] = f2bf(f.y);
      Bs[(bc + 2) * LDB_S + r] = f2bf(f.z);
      Bs[(bc + 3) * LDB_S + r] = f2bf(f.w);
    }
    if (k0 + BK < K) {  // prefetch next B K-slab
      __builtin_prefetch(W + (size_t)(k0 + BK + br) * N + n0 + bc, 0, 0);
    }
    __asm__ __volatile__("s_wait_asynccnt 0x0" ::: "memory");
    __syncthreads();

    // ---- fragments from LDS ----
    FragU af[2], bf[2];
#pragma unroll
    for (int i = 0; i < 2; ++i) {
      const int m = wm * 32 + i * 16 + mlow;
      const int n = wn * 32 + i * 16 + mlow;
#pragma unroll
      for (int v = 0; v < 8; ++v) {
        const float2 p = *(const float2*)&Asf[m * LDA_SF + a_kidx(v, half)];
        af[i].u[v] = pack2(p.x, p.y);
        bf[i].u[v] = *(const unsigned*)&Bs[n * LDB_S + b_kidx(v, half)];
      }
    }
#pragma unroll
    for (int i = 0; i < 2; ++i)
#pragma unroll
      for (int j = 0; j < 2; ++j)
        acc[i][j] = wmma_bf16(af[i].v, bf[j].v, acc[i][j]);
    __syncthreads();
  }

  // ---- epilogue: add bias, store ----
#pragma unroll
  for (int i = 0; i < 2; ++i)
#pragma unroll
    for (int j = 0; j < 2; ++j)
#pragma unroll
      for (int v = 0; v < 8; ++v) {
        const int m = m0 + wm * 32 + i * 16 + v + 8 * half;
        const int n = n0 + wn * 32 + j * 16 + mlow;
        const float val = acc[i][j][v] + bias[n];
        if (OUT_BF16)
          ((unsigned short*)Cout)[(size_t)m * N + n] = f2bf(val);
        else
          ((float*)Cout)[(size_t)m * N + n] = val;
      }
}

// =====================================================================
// RoPE + reorder: mixed bf16 (S, NH, 3, D)  ->
//   q (NH,S,D) bf16 scaled by 1/sqrt(D)  (coeff cancels: reference does
//   q/(sqrt(d)*coeff) then scores*coeff), k (NH,S,D) bf16,
//   vT (NH,D,S) bf16 (transposed so P@V B-fragments are contiguous)
// =====================================================================
__global__ __launch_bounds__(128)
void rope_reorder_kernel(const unsigned short* __restrict__ mixed,
                         const int* __restrict__ pos_ids,
                         unsigned short* __restrict__ qo,
                         unsigned short* __restrict__ ko,
                         unsigned short* __restrict__ vto) {
  const int s = blockIdx.x;
  const int head = blockIdx.y;
  const int d = threadIdx.x;                    // 0..127
  const float qscale = 1.0f / sqrtf((float)HDIM);

  const int pos = pos_ids[s];                   // B==1
  const int i = d & 63;
  const float inv_freq = __expf(-(float)i * (logf(10000.0f) / 64.0f));
  const float fr = (float)pos * inv_freq;
  const float c = __cosf(fr), sn = __sinf(fr);

  const unsigned short* base = mixed + (size_t)s * QKV_N + (size_t)head * (3 * HDIM);
  const int dp = (d < 64) ? d + 64 : d - 64;
  const float sgn = (d < 64) ? -1.0f : 1.0f;

  const float qv = bf2f(base[d]);
  const float qr = bf2f(base[dp]);
  const float kv = bf2f(base[HDIM + d]);
  const float kr = bf2f(base[HDIM + dp]);

  const float qrot = qv * c + sgn * qr * sn;
  const float krot = kv * c + sgn * kr * sn;

  qo[((size_t)head * SEQ + s) * HDIM + d] = f2bf(qrot * qscale);
  ko[((size_t)head * SEQ + s) * HDIM + d] = f2bf(krot);
  vto[((size_t)head * HDIM + d) * SEQ + s] = base[2 * HDIM + d];
}

// =====================================================================
// Flash attention: grid (S/128, NH), 8 waves/block, each wave owns 16
// query rows, streams 32 keys / iteration with online softmax.
//   scores: 8 WMMA (2 key subtiles x 4 K-steps over D=128)
//   P @ V : 8 WMMA (8 d-tiles, K=32 keys)
// Interior iterations are mask-free; the single diagonal iteration is
// peeled with compile-time masking.
// =====================================================================
__global__ __launch_bounds__(256)
void attention_kernel(const unsigned short* __restrict__ qp,
                      const unsigned short* __restrict__ kp,
                      const unsigned short* __restrict__ vtp,
                      const int* __restrict__ layer_id_p,
                      float* __restrict__ ctx) {
  __shared__ float P_s[8][16 * 32];             // per-wave P bounce buffer

  const int lane = threadIdx.x & 31;
  const int wave = __builtin_amdgcn_readfirstlane(threadIdx.x >> 5);
  const int head = blockIdx.y;
  const int qbase = blockIdx.x * 128 + wave * 16;   // wave-scalar
  const float coeff = (float)(*layer_id_p + 1);
  const float maskval = -10000.0f * coeff;
  const int mlow = lane & 15;
  const int half = lane >> 4;

  const v8f vzero = {0.f,0.f,0.f,0.f,0.f,0.f,0.f,0.f};

  // ---- load Q fragments (16 rows x 128 d) : 4 K-steps resident ----
  FragU qf[4];
  const unsigned short* qrow = qp + ((size_t)head * SEQ + (qbase + mlow)) * HDIM;
#pragma unroll
  for (int ks = 0; ks < 4; ++ks)
#pragma unroll
    for (int v = 0; v < 8; ++v)
      qf[ks].u[v] = *(const unsigned*)(qrow + ks * 32 + a_kidx(v, half));

  v8f acc[8];
#pragma unroll
  for (int dt = 0; dt < 8; ++dt) acc[dt] = vzero;
  float mrun[8], lrun[8];
#pragma unroll
  for (int v = 0; v < 8; ++v) { mrun[v] = -3.0e5f; lrun[v] = 0.0f; }

  // One 32-key step of the flash loop; `masked` is a literal at each
  // inlined call site so the diagonal compare/cndmask block folds away
  // in the interior loop.
  auto kv_step = [&](int kb, bool masked) {
    // ---- scores for 2 key subtiles of 16 ----
    v8f s[2]; s[0] = vzero; s[1] = vzero;
#pragma unroll
    for (int t = 0; t < 2; ++t) {
      const unsigned short* krow =
          kp + ((size_t)head * SEQ + (kb + t * 16 + mlow)) * HDIM;
#pragma unroll
      for (int ks = 0; ks < 4; ++ks) {
        FragU kf;
#pragma unroll
        for (int v = 0; v < 8; ++v)
          kf.u[v] = *(const unsigned*)(krow + ks * 32 + b_kidx(v, half));
        s[t] = wmma_bf16(qf[ks].v, kf.v, s[t]);
      }
    }
    // ---- causal mask (diagonal tile only) ----
    if (masked) {
#pragma unroll
      for (int t = 0; t < 2; ++t) {
        const int key = kb + t * 16 + mlow;
#pragma unroll
        for (int v = 0; v < 8; ++v) {
          const int row = qbase + v + 8 * half;
          s[t][v] = (key > row) ? maskval : s[t][v];
        }
      }
    }
    // ---- online softmax (rows live in (v,half); reduce over 16 lanes) ----
#pragma unroll
    for (int v = 0; v < 8; ++v) {
      float tm = fmaxf(s[0][v], s[1][v]);
      tm = fmaxf(tm, __shfl_xor(tm, 1));
      tm = fmaxf(tm, __shfl_xor(tm, 2));
      tm = fmaxf(tm, __shfl_xor(tm, 4));
      tm = fmaxf(tm, __shfl_xor(tm, 8));
      const float mnew = fmaxf(mrun[v], tm);
      const float scale = __expf(mrun[v] - mnew);
      const float p0 = __expf(s[0][v] - mnew);
      const float p1 = __expf(s[1][v] - mnew);
      s[0][v] = p0; s[1][v] = p1;
      float rs = p0 + p1;
      rs += __shfl_xor(rs, 1);
      rs += __shfl_xor(rs, 2);
      rs += __shfl_xor(rs, 4);
      rs += __shfl_xor(rs, 8);
      lrun[v] = lrun[v] * scale + rs;
      mrun[v] = mnew;
#pragma unroll
      for (int dt = 0; dt < 8; ++dt) acc[dt][v] *= scale;
    }
    // ---- transpose P (C-layout f32) -> A-layout bf16 via per-wave LDS ----
#pragma unroll
    for (int t = 0; t < 2; ++t)
#pragma unroll
      for (int v = 0; v < 8; ++v)
        P_s[wave][(v + 8 * half) * 32 + t * 16 + mlow] = s[t][v];
    __asm__ __volatile__("s_wait_dscnt 0" ::: "memory");
    FragU pf;
#pragma unroll
    for (int v = 0; v < 8; ++v) {
      const int kk = a_kidx(v, half);
      pf.u[v] = pack2(P_s[wave][mlow * 32 + kk], P_s[wave][mlow * 32 + kk + 1]);
    }
    // ---- ctx += P (16x32) @ V (32x128) : V read from transposed layout ----
#pragma unroll
    for (int dt = 0; dt < 8; ++dt) {
      FragU vf;
      const int dcol = dt * 16 + mlow;
#pragma unroll
      for (int v = 0; v < 8; ++v)
        vf.u[v] = *(const unsigned*)(vtp + ((size_t)head * HDIM + dcol) * SEQ +
                                     kb + b_kidx(v, half));
      acc[dt] = wmma_bf16(pf.v, vf.v, acc[dt]);
    }
    __asm__ __volatile__("s_wait_dscnt 0" ::: "memory");
  };

  // Interior (mask-free) iterations: kb+31 <= qbase.
  const int kfull = ((qbase + 1) >> 5) << 5;
  for (int kb = 0; kb < kfull; kb += 32) kv_step(kb, false);
  // Exactly one diagonal (masked) iteration remains.
  kv_step(kfull, true);

  // ---- normalize + store ctx (S, H) fp32, head-major columns ----
#pragma unroll
  for (int v = 0; v < 8; ++v) {
    const float rinv = 1.0f / lrun[v];
    const int srow = qbase + v + 8 * half;
#pragma unroll
    for (int dt = 0; dt < 8; ++dt)
      ctx[(size_t)srow * HID + head * HDIM + dt * 16 + mlow] = acc[dt][v] * rinv;
  }
}

// =====================================================================
// Launch
// =====================================================================
extern "C" void kernel_launch(void* const* d_in, const int* in_sizes, int n_in,
                              void* d_out, int out_size, void* d_ws, size_t ws_size,
                              hipStream_t stream) {
  const float* hidden  = (const float*)d_in[0];
  const int*   pos     = (const int*)d_in[1];
  // d_in[2] attention_mask: causal triu — handled analytically
  const float* W_qkv   = (const float*)d_in[3];
  const float* b_qkv   = (const float*)d_in[4];
  const float* W_dense = (const float*)d_in[5];
  const float* b_dense = (const float*)d_in[6];
  const int*   layerid = (const int*)d_in[7];

  char* ws = (char*)d_ws;
  unsigned short* mixed = (unsigned short*)ws;                       // S*3H bf16
  unsigned short* q  = (unsigned short*)(ws + (size_t)SEQ * QKV_N * 2);
  unsigned short* k  = q + (size_t)SEQ * HID;
  unsigned short* vt = k + (size_t)SEQ * HID;
  float* ctx = (float*)(vt + (size_t)SEQ * HID);

  // 1) QKV projection -> bf16 mixed
  gemm_bf16_kernel<true><<<dim3(QKV_N / BN, SEQ / BM), 256, 0, stream>>>(
      hidden, W_qkv, b_qkv, (void*)mixed, SEQ, QKV_N, HID);

  // 2) RoPE + head-major reorder (+ V transpose)
  rope_reorder_kernel<<<dim3(SEQ, NHEAD), 128, 0, stream>>>(
      mixed, pos, q, k, vt);

  // 3) Flash attention per head
  attention_kernel<<<dim3(SEQ / 128, NHEAD), 256, 0, stream>>>(
      q, k, vt, layerid, ctx);

  // 4) Output projection -> fp32 d_out
  gemm_bf16_kernel<false><<<dim3(HID / BN, SEQ / BM), 256, 0, stream>>>(
      ctx, W_dense, b_dense, d_out, SEQ, HID, HID);
}
